// Transformer_13280038879397
// MI455X (gfx1250) — matrix operand order repaired and checked by reading, not compiled, not run on hardware
//
#include <hip/hip_runtime.h>
#include <hip/hip_bf16.h>

typedef __attribute__((ext_vector_type(16))) _Float16 v16h;
typedef __attribute__((ext_vector_type(8)))  float    v8f;
typedef __attribute__((ext_vector_type(4)))  _Float16 h4;
typedef __attribute__((ext_vector_type(4)))  unsigned int u32x4;
typedef __attribute__((ext_vector_type(8)))  int i32x8;
typedef __attribute__((ext_vector_type(4)))  int i32x4;

#define Bn   8
#define Sn   2048
#define Dn   512
#define BSn  (Bn * Sn)          // 16384 rows
#define QSCALE 0.125f           // (512/8)^-0.5
#define LN_EPS 1e-3f

// ---------------------------------------------------------------------------
// WMMA helper: D = A(16x32 f16) * B(32x16 f16) + C(16x16 f32)
// ---------------------------------------------------------------------------
__device__ __forceinline__ v8f wmma16(v16h a, v16h b, v8f c) {
    return __builtin_amdgcn_wmma_f32_16x16x32_f16(
        false, a, false, b, (short)0, c, false, false);
}

// A-fragment (16x32 f16, row-major, leading dim ld). k = 16*(lane/16) + s, same order as B.
__device__ __forceinline__ v16h load_a_frag(const _Float16* p, int ld) {
    const int lane  = threadIdx.x & 31;
    const int m     = lane & 15;
    const int kbase = (lane >> 4) * 16;
    const _Float16* row = p + (size_t)m * ld;
    v16h a;
#pragma unroll
    for (int j = 0; j < 8; ++j)  a[j]     = row[kbase + j];
#pragma unroll
    for (int j = 0; j < 8; ++j)  a[8 + j] = row[kbase + 8 + j];
    return a;
}

// B-fragment (32x16 f16) from N-major memory: element (k,n) = p[n*ld + k].
__device__ __forceinline__ v16h load_bT_frag(const _Float16* p, int ld) {
    const int lane  = threadIdx.x & 31;
    const int n     = lane & 15;
    const int kbase = (lane >> 4) * 16;
    const _Float16* row = p + (size_t)n * ld + kbase;
    v16h b;
#pragma unroll
    for (int j = 0; j < 16; ++j) b[j] = row[j];
    return b;
}

// ---------------------------------------------------------------------------
// CDNA5 async global->LDS staging (ASYNCcnt-tracked)
// ---------------------------------------------------------------------------
__device__ __forceinline__ uint32_t lds_off32(const void* p) {
    return (uint32_t)(uintptr_t)p;   // LDS aperture: flat addr[31:0] == LDS offset
}
__device__ __forceinline__ void async_copy_b128(uint32_t lds_byte_off, const void* gsrc) {
    asm volatile("global_load_async_to_lds_b128 %0, %1, off"
                 :: "v"(lds_byte_off), "v"(gsrc) : "memory");
}
__device__ __forceinline__ void wait_async0() {
    asm volatile("s_wait_asynccnt 0x0" ::: "memory");
}

// ---------------------------------------------------------------------------
// CDNA5 Tensor Data Mover: 2D tile (f16) global -> LDS with row padding.
// D# fields per ISA 8.3/8.4. One wave issues; completion via TENSORcnt.
// 6-arg builtin variant: (g0, g1, g2, g3, g4, cpol).
// ---------------------------------------------------------------------------
__device__ __forceinline__ void tdm_load_2d_f16(
    uint32_t lds_addr, const void* gaddr,
    uint32_t tile_w_halves, uint32_t tile_h_rows,
    uint32_t tensor_w_halves, uint32_t tensor_h_rows,
    uint32_t row_stride_halves, uint32_t pad_amount_dw /*0=no pad*/)
{
    const unsigned long long ga = (unsigned long long)(uintptr_t)gaddr;
    u32x4 g0;
    g0[0] = 1u;                                  // count=1 (valid descriptor)
    g0[1] = lds_addr;                            // lds_addr [63:32]
    g0[2] = (unsigned int)ga;                    // global_addr lo
    g0[3] = (unsigned int)(ga >> 32) | (2u << 30); // global_addr hi | type=2
    unsigned int flags = (1u << 16);             // data_size = 2 bytes
    if (pad_amount_dw)
        flags |= (1u << 20)                      // pad_enable
               | (7u << 22)                      // pad_interval: 256 DWORDs / row
               | ((pad_amount_dw - 1u) << 25);   // pad_amount DWORDs
    i32x8 g1;
    g1[0] = (int)flags;                          // wg_mask=0, flags
    g1[1] = (int)(tensor_w_halves << 16);        // tensor_dim0 lo16 @ [63:48]
    g1[2] = (int)((tensor_w_halves >> 16) | (tensor_h_rows << 16)); // dim0 hi | dim1 lo
    g1[3] = (int)((tensor_h_rows >> 16) | (tile_w_halves << 16));   // dim1 hi | tile_dim0
    g1[4] = (int)tile_h_rows;                    // tile_dim1 (tile_dim2 = 0)
    g1[5] = (int)row_stride_halves;              // tensor_dim0_stride lo32
    g1[6] = 0;
    g1[7] = 0;
    i32x4 z4 = {0, 0, 0, 0};
    i32x8 z8 = {0, 0, 0, 0, 0, 0, 0, 0};
    __builtin_amdgcn_tensor_load_to_lds(g0, g1, z4, z4, z8, 0);
}
__device__ __forceinline__ void wait_tensor0() {
    __builtin_amdgcn_s_wait_tensorcnt((short)0);
}

// ---------------------------------------------------------------------------
// fp32 -> f16 conversion (vectorized), fused convert+transpose for weights
// ---------------------------------------------------------------------------
__global__ __launch_bounds__(256) void cvt_f32_f16_v4(const float* __restrict__ src,
                                                      _Float16* __restrict__ dst, int n4) {
    int i = blockIdx.x * blockDim.x + threadIdx.x;
    int stride = gridDim.x * blockDim.x;
    for (; i < n4; i += stride) {
        float4 f = ((const float4*)src)[i];
        h4 o = { (_Float16)f.x, (_Float16)f.y, (_Float16)f.z, (_Float16)f.w };
        ((h4*)dst)[i] = o;
    }
}

__global__ __launch_bounds__(256) void cvt_transpose_512(const float* __restrict__ src,
                                                         _Float16* __restrict__ dst) {
    const int i = blockIdx.x * 256 + threadIdx.x;
    const int k = i >> 9, n = i & 511;
    dst[(size_t)n * Dn + k] = (_Float16)src[i];
}

// ---------------------------------------------------------------------------
// Fused QKV projection, 32-row M-tiles (2 per wave -> each B-frag feeds 2 WMMA).
// grid = (BSn/32, Dn/64), block = 128.
// ---------------------------------------------------------------------------
__global__ __launch_bounds__(128) void qkv_gemm(
    const _Float16* __restrict__ xh,
    const _Float16* __restrict__ wqT, const _Float16* __restrict__ wkT,
    const _Float16* __restrict__ wvT,
    const float* __restrict__ bq, const float* __restrict__ bk,
    const float* __restrict__ bv,
    _Float16* __restrict__ q, _Float16* __restrict__ k, _Float16* __restrict__ v)
{
    __shared__ _Float16 xt[32 * Dn];                    // 32 KB
    const int mt   = blockIdx.x;
    const int tid  = threadIdx.x;
    const int wave = tid >> 5;
    const int lane = tid & 31;

    const char* xsrc = (const char*)(xh + (size_t)mt * 32 * Dn);
    for (int i = tid; i < 2048; i += 128)
        async_copy_b128(lds_off32((const char*)xt + i * 16), xsrc + i * 16);
    wait_async0();
    __syncthreads();

    const int n0 = blockIdx.y * 64 + wave * 16;
    v8f aq0 = {}, aq1 = {}, ak0 = {}, ak1 = {}, av0 = {}, av1 = {};
#pragma unroll
    for (int kk = 0; kk < Dn; kk += 32) {
        v16h a0 = load_a_frag(xt + kk, Dn);
        v16h a1 = load_a_frag(xt + 16 * Dn + kk, Dn);
        v16h bf;
        bf = load_bT_frag(wqT + (size_t)n0 * Dn + kk, Dn);
        aq0 = wmma16(a0, bf, aq0); aq1 = wmma16(a1, bf, aq1);
        bf = load_bT_frag(wkT + (size_t)n0 * Dn + kk, Dn);
        ak0 = wmma16(a0, bf, ak0); ak1 = wmma16(a1, bf, ak1);
        bf = load_bT_frag(wvT + (size_t)n0 * Dn + kk, Dn);
        av0 = wmma16(a0, bf, av0); av1 = wmma16(a1, bf, av1);
    }

    const int n     = n0 + (lane & 15);
    const int mbase = mt * 32 + (lane >> 4) * 8;
    const float bqv = bq[n], bkv = bk[n], bvv = bv[n];
#pragma unroll
    for (int r = 0; r < 8; ++r) {
        size_t o = (size_t)(mbase + r) * Dn + n;
        q[o] = (_Float16)(fmaxf(aq0[r] + bqv, 0.f) * QSCALE);
        k[o] = (_Float16)fmaxf(ak0[r] + bkv, 0.f);
        v[o] = (_Float16)fmaxf(av0[r] + bvv, 0.f);
        o += (size_t)16 * Dn;
        q[o] = (_Float16)(fmaxf(aq1[r] + bqv, 0.f) * QSCALE);
        k[o] = (_Float16)fmaxf(ak1[r] + bkv, 0.f);
        v[o] = (_Float16)fmaxf(av1[r] + bvv, 0.f);
    }
}

// ---------------------------------------------------------------------------
// Flash attention, full D=512 contraction. q/k tiles staged by the TDM.
// grid=(Sn/64, Bn), block=256 (8 waves). Wave w: m-tile w&3, value half w>>2.
// ---------------------------------------------------------------------------
#define QLD 520
#define KLD 520
#define VLD 48
#define ATTN_LDS_BYTES ((64 * QLD + 32 * KLD + Dn * VLD + 8 * 16 * 32) * 2)

__global__ __launch_bounds__(256) void attn_kernel(
    const _Float16* __restrict__ qg, const _Float16* __restrict__ kg,
    const _Float16* __restrict__ vg, _Float16* __restrict__ ctx)
{
    extern __shared__ _Float16 smem[];
    _Float16* qt  = smem;                       // 64 x QLD
    _Float16* kt  = qt + 64 * QLD;              // 32 x KLD
    _Float16* vtT = kt + 32 * KLD;              // 512 x VLD (dim-major)
    _Float16* pb  = vtT + Dn * VLD;             // 8 waves x (16 x 32)

    const int bat  = blockIdx.y;
    const int mt   = blockIdx.x;
    const int tid  = threadIdx.x;
    const int wave = tid >> 5;
    const int lane = tid & 31;
    const int mw   = wave & 3;
    const int hv   = wave >> 2;

    // TDM: stage the 64x512 query tile into padded LDS rows (pad 4 DWORDs/row)
    if (wave == 0)
        tdm_load_2d_f16(lds_off32(qt), qg + ((size_t)bat * Sn + mt * 64) * Dn,
                        Dn, 64, Dn, 64, Dn, (QLD - Dn) / 2);

    v8f acc[16];
#pragma unroll
    for (int t = 0; t < 16; ++t) acc[t] = (v8f){};
    float mrow[8], lrow[8];
#pragma unroll
    for (int r = 0; r < 8; ++r) { mrow[r] = -1e30f; lrow[r] = 0.f; }

    _Float16* mypb = pb + wave * (16 * 32);

    for (int it = 0; it < Sn / 32; ++it) {
        __syncthreads();   // prior iteration done reading kt/vtT
        const size_t kvbase = ((size_t)bat * Sn + it * 32) * Dn;
        // TDM: k tile (32x512 -> padded rows)
        if (wave == 0)
            tdm_load_2d_f16(lds_off32(kt), kg + kvbase,
                            Dn, 32, Dn, 32, Dn, (KLD - Dn) / 2);
        // v tile: transpose to dim-major while staging
        const _Float16* vsrc = vg + kvbase;
        for (int i = tid; i < 2048; i += 256) {
            const int key = i >> 6, c = i & 63;
            const uint4 dv = *(const uint4*)(vsrc + (size_t)key * Dn + c * 8);
            const _Float16* hp = (const _Float16*)&dv;
#pragma unroll
            for (int j = 0; j < 8; ++j) vtT[(c * 8 + j) * VLD + key] = hp[j];
        }
        wait_tensor0();      // wave 0 waits its TDM ops; no-op elsewhere
        __syncthreads();

        // ---- scores: S[16 x 32] = q_tile @ k_tile^T ----
        v8f s0 = {}, s1 = {};
#pragma unroll
        for (int kc = 0; kc < Dn; kc += 32) {
            v16h af = load_a_frag(qt + (size_t)mw * 16 * QLD + kc, QLD);
            v16h b0 = load_bT_frag(kt + kc, KLD);
            s0 = wmma16(af, b0, s0);
            v16h b1 = load_bT_frag(kt + (size_t)16 * KLD + kc, KLD);
            s1 = wmma16(af, b1, s1);
        }

        // ---- online softmax ----
        float alpha[8];
#pragma unroll
        for (int r = 0; r < 8; ++r) {
            float x = fmaxf(s0[r], s1[r]);
#pragma unroll
            for (int m = 8; m >= 1; m >>= 1) x = fmaxf(x, __shfl_xor(x, m, 32));
            const float mnew = fmaxf(mrow[r], x);
            alpha[r] = __expf(mrow[r] - mnew);
            mrow[r]  = mnew;

            const float p0 = __expf(s0[r] - mnew);
            const float p1 = __expf(s1[r] - mnew);
            const int prow = r + (lane >> 4) * 8;
            mypb[prow * 32 + (lane & 15)]      = (_Float16)p0;
            mypb[prow * 32 + 16 + (lane & 15)] = (_Float16)p1;
            float s = p0 + p1;
#pragma unroll
            for (int m = 8; m >= 1; m >>= 1) s += __shfl_xor(s, m, 32);
            lrow[r] = lrow[r] * alpha[r] + s;
        }

#pragma unroll
        for (int t = 0; t < 16; ++t)
#pragma unroll
            for (int r = 0; r < 8; ++r) acc[t][r] *= alpha[r];

        // ---- ctx += P(16x32) @ V(32 x 256-slice) ----
        v16h pf = load_a_frag(mypb, 32);
#pragma unroll
        for (int t = 0; t < 16; ++t) {
            v16h bf = load_bT_frag(vtT + (size_t)(hv * 256 + t * 16) * VLD, VLD);
            acc[t] = wmma16(pf, bf, acc[t]);
        }
    }

#pragma unroll
    for (int r = 0; r < 8; ++r) {
        const float inv = 1.0f / lrow[r];
        const int row = bat * Sn + mt * 64 + mw * 16 + r + (lane >> 4) * 8;
#pragma unroll
        for (int t = 0; t < 16; ++t) {
            const int col = hv * 256 + t * 16 + (lane & 15);
            ctx[(size_t)row * Dn + col] = (_Float16)(acc[t][r] * inv);
        }
    }
}

// ---------------------------------------------------------------------------
// Output projection, 32-row M-tiles: pre = relu(ctx @ Wo + bo), fp32 out.
// grid = (BSn/32, Dn/64), block = 128.
// ---------------------------------------------------------------------------
__global__ __launch_bounds__(128) void out_gemm(
    const _Float16* __restrict__ ch, const _Float16* __restrict__ woT,
    const float* __restrict__ bo, float* __restrict__ pre)
{
    __shared__ _Float16 ct[32 * Dn];
    const int mt   = blockIdx.x;
    const int tid  = threadIdx.x;
    const int wave = tid >> 5;
    const int lane = tid & 31;

    const char* csrc = (const char*)(ch + (size_t)mt * 32 * Dn);
    for (int i = tid; i < 2048; i += 128)
        async_copy_b128(lds_off32((const char*)ct + i * 16), csrc + i * 16);
    wait_async0();
    __syncthreads();

    const int n0 = blockIdx.y * 64 + wave * 16;
    v8f a0 = {}, a1 = {};
#pragma unroll
    for (int kk = 0; kk < Dn; kk += 32) {
        v16h f0 = load_a_frag(ct + kk, Dn);
        v16h f1 = load_a_frag(ct + 16 * Dn + kk, Dn);
        v16h bf = load_bT_frag(woT + (size_t)n0 * Dn + kk, Dn);
        a0 = wmma16(f0, bf, a0);
        a1 = wmma16(f1, bf, a1);
    }

    const int n     = n0 + (lane & 15);
    const int mbase = mt * 32 + (lane >> 4) * 8;
    const float bv  = bo[n];
#pragma unroll
    for (int r = 0; r < 8; ++r) {
        pre[(size_t)(mbase + r) * Dn + n]      = fmaxf(a0[r] + bv, 0.f);
        pre[(size_t)(mbase + 16 + r) * Dn + n] = fmaxf(a1[r] + bv, 0.f);
    }
}

// ---------------------------------------------------------------------------
// LayerNorm over last dim (512), eps = 1e-3. One block (128 thr) per row.
// ---------------------------------------------------------------------------
__global__ __launch_bounds__(128) void layernorm_kernel(
    const float* __restrict__ pre, const float* __restrict__ gamma,
    const float* __restrict__ beta, float* __restrict__ out)
{
    __shared__ float red[8];
    const int row  = blockIdx.x;
    const int tid  = threadIdx.x;
    const int wid  = tid >> 5;
    const int lane = tid & 31;
    const float* p = pre + (size_t)row * Dn;

    float s = 0.f, ss = 0.f;
    for (int i = tid; i < Dn; i += 128) { float v = p[i]; s += v; ss += v * v; }
#pragma unroll
    for (int m = 16; m >= 1; m >>= 1) { s += __shfl_xor(s, m, 32); ss += __shfl_xor(ss, m, 32); }
    if (lane == 0) { red[wid] = s; red[4 + wid] = ss; }
    __syncthreads();
    if (tid == 0) {
        red[0] = red[0] + red[1] + red[2] + red[3];
        red[4] = red[4] + red[5] + red[6] + red[7];
    }
    __syncthreads();
    const float mean = red[0] * (1.0f / Dn);
    const float var  = red[4] * (1.0f / Dn) - mean * mean;
    const float inv  = rsqrtf(var + LN_EPS);
    for (int i = tid; i < Dn; i += 128)
        out[(size_t)row * Dn + i] = (p[i] - mean) * inv * gamma[i] + beta[i];
}

// ---------------------------------------------------------------------------
// launch
// ---------------------------------------------------------------------------
extern "C" void kernel_launch(void* const* d_in, const int* in_sizes, int n_in,
                              void* d_out, int out_size, void* d_ws, size_t ws_size,
                              hipStream_t stream) {
    const float* x     = (const float*)d_in[0];
    const float* Wq    = (const float*)d_in[1];
    const float* bq    = (const float*)d_in[2];
    const float* Wk    = (const float*)d_in[3];
    const float* bk    = (const float*)d_in[4];
    const float* Wv    = (const float*)d_in[5];
    const float* bv    = (const float*)d_in[6];
    const float* Wo    = (const float*)d_in[7];
    const float* bo    = (const float*)d_in[8];
    const float* gamma = (const float*)d_in[9];
    const float* beta  = (const float*)d_in[10];

    char* w = (char*)d_ws;
    size_t o = 0;
    _Float16* xh  = (_Float16*)(w + o); o += (size_t)BSn * Dn * 2;
    _Float16* wqT = (_Float16*)(w + o); o += (size_t)Dn * Dn * 2;
    _Float16* wkT = (_Float16*)(w + o); o += (size_t)Dn * Dn * 2;
    _Float16* wvT = (_Float16*)(w + o); o += (size_t)Dn * Dn * 2;
    _Float16* woT = (_Float16*)(w + o); o += (size_t)Dn * Dn * 2;
    _Float16* qh  = (_Float16*)(w + o); o += (size_t)BSn * Dn * 2;
    _Float16* kh  = (_Float16*)(w + o); o += (size_t)BSn * Dn * 2;
    _Float16* vh  = (_Float16*)(w + o); o += (size_t)BSn * Dn * 2;
    _Float16* ch  = (_Float16*)(w + o); o += (size_t)BSn * Dn * 2;
    float*    pre = (float*)(w + o);    o += (size_t)BSn * Dn * 4;

    cvt_f32_f16_v4<<<2048, 256, 0, stream>>>(x, xh, BSn * Dn / 4);
    cvt_transpose_512<<<1024, 256, 0, stream>>>(Wq, wqT);
    cvt_transpose_512<<<1024, 256, 0, stream>>>(Wk, wkT);
    cvt_transpose_512<<<1024, 256, 0, stream>>>(Wv, wvT);
    cvt_transpose_512<<<1024, 256, 0, stream>>>(Wo, woT);

    qkv_gemm<<<dim3(BSn / 32, Dn / 64), 128, 0, stream>>>(
        xh, wqT, wkT, wvT, bq, bk, bv, qh, kh, vh);

    (void)hipFuncSetAttribute((const void*)attn_kernel,
                              hipFuncAttributeMaxDynamicSharedMemorySize, ATTN_LDS_BYTES);
    attn_kernel<<<dim3(Sn / 64, Bn), 256, ATTN_LDS_BYTES, stream>>>(qh, kh, vh, ch);

    out_gemm<<<dim3(BSn / 32, Dn / 64), 128, 0, stream>>>(ch, woT, bo, pre);

    layernorm_kernel<<<BSn, 128, 0, stream>>>(pre, gamma, beta, (float*)d_out);
}
